// AutoCorrelationLayer_65438121722674
// MI455X (gfx1250) — matrix operand order repaired and checked by reading, not compiled, not run on hardware
//
#include <hip/hip_runtime.h>
#include <hip/hip_bf16.h>
#include <stdint.h>

// ---------------------------------------------------------------------------
// Types for CDNA5 WMMA (wave32). v16bf/v8f match the probe-confirmed builtin
// signature of __builtin_amdgcn_wmma_f32_16x16x32_bf16.
// ---------------------------------------------------------------------------
typedef __bf16 bf16_t;
typedef __attribute__((ext_vector_type(16))) __bf16 v16bf;
typedef __attribute__((ext_vector_type(8)))  float  v8f;

struct Frag { uint4 lo, hi; };   // 32 bytes == one 16-element bf16 WMMA operand

__device__ __forceinline__ bf16_t f2bf(float f) {
    unsigned int u = __builtin_bit_cast(unsigned int, f);
    u = (u + 0x7FFFu + ((u >> 16) & 1u)) >> 16;       // round-to-nearest-even
    unsigned short s = (unsigned short)u;
    return __builtin_bit_cast(bf16_t, s);
}
__device__ __forceinline__ float bf2f(bf16_t h) {
    unsigned short s = __builtin_bit_cast(unsigned short, h);
    unsigned int u = ((unsigned int)s) << 16;
    return __builtin_bit_cast(float, u);
}

// ---------------------------------------------------------------------------
// fp32 -> bf16 conversion (vectorized: float4 in, 4x bf16 out)
// ---------------------------------------------------------------------------
__global__ void cvt_f32_to_bf16(const float4* __restrict__ src,
                                uint2* __restrict__ dst, long n4) {
    long i = (long)blockIdx.x * blockDim.x + threadIdx.x;
    long stride = (long)gridDim.x * blockDim.x;
    for (; i < n4; i += stride) {
        float4 v = src[i];
        unsigned int a = __builtin_bit_cast(unsigned short, f2bf(v.x));
        unsigned int b = __builtin_bit_cast(unsigned short, f2bf(v.y));
        unsigned int c = __builtin_bit_cast(unsigned short, f2bf(v.z));
        unsigned int d = __builtin_bit_cast(unsigned short, f2bf(v.w));
        uint2 o; o.x = a | (b << 16); o.y = c | (d << 16);
        dst[i] = o;
    }
}

// ---------------------------------------------------------------------------
// Generic bf16 WMMA GEMM:  C[M,N] = A[M,K] * B + (bias)
//   BT=true : B is W[N,K] row-major (contraction over K)  -> x @ W^T
//   BT=false: B is [K,N] row-major with row stride ldb (batched aggregation)
// Workgroup: 256 threads = 8 waves, block tile 128(M) x 128(N), K-step 32,
// double-buffered LDS (one barrier per K-step). Wave w -> (mw=w&3, nw=w>>2)
// owns a 32x64 tile = 2x4 v_wmma_f32_16x16x32_bf16 per K-step (8 WMMA vs
// 12 ds_load_b128 -> ~2x the WMMA density of a 32x32 wave tile).
// LDS: As [128][32] row-major, Bs canonical [n][k] (128x32); fragments are
// two 16B LDS reads each per the ISA bf16 VGPR layouts:
//   A 16x32: lanes 0-15 M=0..15 (K 0..7,16..23), lanes 16-31 (K 8..15,24..31)
//   B 32x16: lanes 0-15 N (K 0..7,16..23), lanes 16-31 same N (K 8..15,24..31)
// ---------------------------------------------------------------------------
#define BM 128
#define BN 128
#define KSZ 32

template <bool BT, bool OUT_F32, bool BIAS>
__global__ __launch_bounds__(256) void gemm_wmma(
    const bf16_t* __restrict__ A, const bf16_t* __restrict__ Bm,
    const float* __restrict__ bias, void* __restrict__ C,
    int M, int N, int K, int lda, int ldb, int ldc,
    long sA, long sBb, long sBh, long sCb, long sCh, int Hh)
{
    __shared__ __align__(16) bf16_t As[2][BM * KSZ];   // 2 x 8 KB
    __shared__ __align__(16) bf16_t Bs[2][BN * KSZ];   // 2 x 8 KB

    const int tid  = threadIdx.x;
    const int lane = tid & 31;
    const int wave = tid >> 5;
    const int mw   = wave & 3;       // 4 waves along M
    const int nw   = wave >> 2;      // 2 waves along N

    const int bz = blockIdx.z;
    const int bb = bz / Hh, bh = bz - bb * Hh;
    const bf16_t* Ab = A + (long)bz * sA + (long)blockIdx.y * BM * lda;
    const bf16_t* Bb = Bm + (long)bb * sBb + (long)bh * sBh;
    const long cOff  = (long)bb * sCb + (long)bh * sCh;

    v8f acc[2][4] = {};

    // staging thread mapping (16 bf16 = two uint4 per thread for A and B)
    const int smA  = tid >> 1;            // A row 0..127
    const int skA  = (tid & 1) * 16;      // A k-chunk
    const int snB  = tid >> 1;            // B row (BT) 0..127
    const int skB  = (tid & 1) * 16;
    const int tkB  = tid & 31;            // B k (transpose path)
    const int tnB  = (tid >> 5) * 16;     // B n-chunk (transpose path)

    // fragment addressing
    const int am  = mw * 32 + (lane & 15);     // A fragment row in tile
    const int akw = (lane >> 4) * 4;           // A word offset (K half)
    const int bn  = nw * 64 + (lane & 15);     // B fragment col in tile
    const int bkw = (lane >> 4) * 4;           // B word offset (K half)

    auto stage = [&](int buf, int k0) {
        // ---- A tile: 128x32, 2 threads/row, 2x 16B vector loads ----
        {
            const uint4* g = (const uint4*)(Ab + (long)smA * lda + k0 + skA);
            uint4* s = (uint4*)(&As[buf][smA * KSZ + skA]);
            s[0] = g[0];
            s[1] = g[1];
            if (k0 + 4 * KSZ < K)   // hide HBM first-touch latency
                __builtin_prefetch(Ab + (long)smA * lda + k0 + 4 * KSZ + skA, 0, 0);
        }
        // ---- B tile into canonical [n][k] layout ----
        if (BT) {   // B = W[N,K] row-major: rows already [n][k]
            const uint4* g = (const uint4*)
                (Bb + (long)(blockIdx.x * BN + snB) * ldb + k0 + skB);
            uint4* s = (uint4*)(&Bs[buf][snB * KSZ + skB]);
            s[0] = g[0];
            s[1] = g[1];
            if (k0 + 4 * KSZ < K)
                __builtin_prefetch(Bb + (long)(blockIdx.x * BN + snB) * ldb
                                       + k0 + 4 * KSZ + skB, 0, 0);
        } else {    // B = [K,N] row-major: transpose on stage
            const uint4* g = (const uint4*)
                (Bb + (long)(k0 + tkB) * ldb + blockIdx.x * BN + tnB);
            __align__(16) bf16_t tmp[16];
            *(uint4*)(tmp)     = g[0];
            *(uint4*)(tmp + 8) = g[1];
            #pragma unroll
            for (int i = 0; i < 16; ++i)
                Bs[buf][(tnB + i) * KSZ + tkB] = tmp[i];
        }
    };

    const int nsteps = K / KSZ;
    stage(0, 0);
    __syncthreads();

    for (int s = 0; s < nsteps; ++s) {
        const int cur = s & 1;
        if (s + 1 < nsteps) stage(cur ^ 1, (s + 1) * KSZ);

        const unsigned int* As32 = (const unsigned int*)As[cur];
        const unsigned int* Bs32 = (const unsigned int*)Bs[cur];

        Frag fa[2], fb[4];
        #pragma unroll
        for (int r = 0; r < 2; ++r) {
            int base = (am + r * 16) * (KSZ / 2) + akw;
            fa[r].lo = *(const uint4*)(As32 + base);
            fa[r].hi = *(const uint4*)(As32 + base + 8);
        }
        #pragma unroll
        for (int c = 0; c < 4; ++c) {
            int base = (bn + c * 16) * (KSZ / 2) + bkw;
            fb[c].lo = *(const uint4*)(Bs32 + base);
            fb[c].hi = *(const uint4*)(Bs32 + base + 8);
        }

        #pragma unroll
        for (int r = 0; r < 2; ++r) {
            v16bf av = __builtin_bit_cast(v16bf, fa[r]);
            #pragma unroll
            for (int c = 0; c < 4; ++c) {
                v16bf bv = __builtin_bit_cast(v16bf, fb[c]);
                acc[r][c] = __builtin_amdgcn_wmma_f32_16x16x32_bf16(
                    false, av, false, bv, (short)0, acc[r][c], false, false);
            }
        }
        __syncthreads();   // covers RAW on next buffer + WAR on this buffer
    }

    // ---- epilogue: C/D f32 layout: VGPR v -> M = v + 8*(lane>=16), N = lane&15
    const int mTile = blockIdx.y * BM + mw * 32;
    const int nTile = blockIdx.x * BN + nw * 64;
    #pragma unroll
    for (int r = 0; r < 2; ++r) {
        #pragma unroll
        for (int c = 0; c < 4; ++c) {
            int n = nTile + c * 16 + (lane & 15);
            float badd = BIAS ? bias[n] : 0.0f;
            int mBase = mTile + r * 16 + ((lane >> 4) << 3);
            #pragma unroll
            for (int v = 0; v < 8; ++v) {
                float val = acc[r][c][v] + badd;
                long idx = cOff + (long)(mBase + v) * ldc + n;
                if (OUT_F32) ((float*)C)[idx] = val;
                else         ((bf16_t*)C)[idx] = f2bf(val);
            }
        }
    }
}

// ---------------------------------------------------------------------------
// Circular cross-correlation over head dim (size 256) + softmax, one wave32
// per (b,h,l) row. k is duplicated in LDS (512 floats) to avoid the modulo.
// corr[tau] = sum_t q[t] * k[(t+tau) % 256]; softmax over tau (temperature/h).
// ---------------------------------------------------------------------------
__global__ __launch_bounds__(256) void corr_softmax(
    const bf16_t* __restrict__ Qp, const bf16_t* __restrict__ Kp,
    const float* __restrict__ temp, bf16_t* __restrict__ Corr)
{
    __shared__ float qs[8][256];
    __shared__ float ks2[8][512];
    const int lane = threadIdx.x & 31;
    const int wave = threadIdx.x >> 5;
    const long row = (long)blockIdx.x * 8 + wave;   // row = ((b*H)+h)*L + l
    const int l = (int)(row & 255);
    const int h = (int)((row >> 8) & 7);
    const int b = (int)(row >> 11);

    const bf16_t* q = Qp + ((long)(b * 256 + l)) * 2048 + h * 256;
    const bf16_t* k = Kp + ((long)(b * 256 + l)) * 2048 + h * 256;
    #pragma unroll
    for (int i = 0; i < 8; ++i) {
        int t = i * 32 + lane;
        float qv = bf2f(q[t]);
        float kv = bf2f(k[t]);
        qs[wave][t] = qv;
        ks2[wave][t] = kv;
        ks2[wave][t + 256] = kv;
    }
    __syncthreads();

    const float scale = 1.0f / temp[h];
    float c[8];
    #pragma unroll
    for (int j = 0; j < 8; ++j) c[j] = 0.0f;
    for (int t0 = 0; t0 < 256; t0 += 8) {
        float qr[8];
        #pragma unroll
        for (int tt = 0; tt < 8; ++tt) qr[tt] = qs[wave][t0 + tt];
        #pragma unroll
        for (int j = 0; j < 8; ++j) {
            int tau = j * 32 + lane;
            float s = c[j];
            #pragma unroll
            for (int tt = 0; tt < 8; ++tt)
                s += qr[tt] * ks2[wave][t0 + tt + tau];
            c[j] = s;
        }
    }

    float mx = -3.4e38f;
    #pragma unroll
    for (int j = 0; j < 8; ++j) { c[j] *= scale; mx = fmaxf(mx, c[j]); }
    for (int off = 16; off > 0; off >>= 1) mx = fmaxf(mx, __shfl_xor(mx, off, 32));
    float sum = 0.0f;
    #pragma unroll
    for (int j = 0; j < 8; ++j) { c[j] = __expf(c[j] - mx); sum += c[j]; }
    for (int off = 16; off > 0; off >>= 1) sum += __shfl_xor(sum, off, 32);
    const float inv = 1.0f / sum;
    #pragma unroll
    for (int j = 0; j < 8; ++j) {
        int tau = j * 32 + lane;
        Corr[row * 256 + tau] = f2bf(c[j] * inv);
    }
}

// ---------------------------------------------------------------------------
// Launcher
// ---------------------------------------------------------------------------
extern "C" void kernel_launch(void* const* d_in, const int* in_sizes, int n_in,
                              void* d_out, int out_size, void* d_ws, size_t ws_size,
                              hipStream_t stream) {
    (void)in_sizes; (void)n_in; (void)out_size; (void)ws_size;
    const float* queries = (const float*)d_in[0];
    const float* keys    = (const float*)d_in[1];
    const float* values  = (const float*)d_in[2];
    const float* Wq = (const float*)d_in[3];
    const float* bq = (const float*)d_in[4];
    const float* Wk = (const float*)d_in[5];
    const float* bk = (const float*)d_in[6];
    const float* Wv = (const float*)d_in[7];
    const float* bv = (const float*)d_in[8];
    const float* Wo = (const float*)d_in[9];
    const float* bo = (const float*)d_in[10];
    const float* temp = (const float*)d_in[11];

    const int  B = 32, L = 256, D = 2048, H = 8, DH = 256;
    const long MN = (long)B * L;        // 8192 rows
    const long XE = MN * D;             // 16,777,216 activation elements
    const long WE = (long)D * D;        // 4,194,304 weight elements

    // Workspace layout (bytes; ~200 MB total). XA doubles as input staging and
    // (after V projection) as the aggregation output.
    char* ws = (char*)d_ws;
    bf16_t* XA  = (bf16_t*)(ws);                    // 33.5 MB: X stage / Agg
    bf16_t* WQb = (bf16_t*)(ws + 33554432);         //  8.4 MB
    bf16_t* WKb = (bf16_t*)(ws + 41943040);
    bf16_t* WVb = (bf16_t*)(ws + 50331648);
    bf16_t* WOb = (bf16_t*)(ws + 58720256);
    bf16_t* QP  = (bf16_t*)(ws + 67108864);         // 33.5 MB
    bf16_t* KP  = (bf16_t*)(ws + 100663296);
    bf16_t* VP  = (bf16_t*)(ws + 134217728);
    bf16_t* CR  = (bf16_t*)(ws + 167772160);        // 33.5 MB corr (bf16)

    dim3 blk(256);
    auto cvt = [&](const float* s, bf16_t* dptr, long n) {
        long n4 = n / 4;
        int nb = (int)((n4 + 255) / 256);
        if (nb > 16384) nb = 16384;
        cvt_f32_to_bf16<<<nb, blk, 0, stream>>>((const float4*)s, (uint2*)dptr, n4);
    };

    // 1) weights to bf16 (once)
    cvt(Wq, WQb, WE); cvt(Wk, WKb, WE); cvt(Wv, WVb, WE); cvt(Wo, WOb, WE);

    dim3 gProj(D / BN, (unsigned)(MN / BM), 1);     // (16, 64, 1)

    // 2) projections: Q, K, V  (x @ W^T + b), bf16 outputs
    cvt(queries, XA, XE);
    gemm_wmma<true,  false, true ><<<gProj, blk, 0, stream>>>(
        XA, WQb, bq, QP, (int)MN, D, D, D, D, D, 0, 0, 0, 0, 0, 1);
    cvt(keys, XA, XE);
    gemm_wmma<true,  false, true ><<<gProj, blk, 0, stream>>>(
        XA, WKb, bk, KP, (int)MN, D, D, D, D, D, 0, 0, 0, 0, 0, 1);
    cvt(values, XA, XE);
    gemm_wmma<true,  false, true ><<<gProj, blk, 0, stream>>>(
        XA, WVb, bv, VP, (int)MN, D, D, D, D, D, 0, 0, 0, 0, 0, 1);

    // 3) circular correlation over head dim + softmax
    corr_softmax<<<(B * H * L) / 8, blk, 0, stream>>>(QP, KP, temp, CR);

    // 4) time-delay aggregation: 256 batched [L x DH x S] GEMMs
    //    Agg[b,l,h*DH+d'] = sum_s corr[b,h,l,s] * Vproj[b,s,h*DH+d']
    dim3 gAgg(DH / BN, L / BM, B * H);              // (2, 2, 256)
    gemm_wmma<false, false, false><<<gAgg, blk, 0, stream>>>(
        CR, VP, nullptr, XA, L, DH, DH, DH, D, D,
        (long)L * DH, (long)L * D, (long)DH, (long)L * D, (long)DH, H);

    // 5) output projection (fp32 result straight to d_out)
    gemm_wmma<true,  true,  true ><<<gProj, blk, 0, stream>>>(
        XA, WOb, bo, (float*)d_out, (int)MN, D, D, D, D, D, 0, 0, 0, 0, 0, 1);
}
